// GNN_89885075570707
// MI455X (gfx1250) — compile-verified, matrix operand order checked by reading, not run on hardware
//
#include <hip/hip_runtime.h>

#define NN 40000
#define NE 640000
#define EMB 128
#define TOT (NN * EMB)            // 5,120,000
#define EPS 1e-5f

typedef _Float16 v16h __attribute__((ext_vector_type(16)));
typedef _Float16 v8h  __attribute__((ext_vector_type(8)));
typedef float    v8f  __attribute__((ext_vector_type(8)));

union Frag { v16h v; v8h h[2]; };

// ---------------------------------------------------------------- utilities
__global__ __launch_bounds__(256) void k_zero(float* __restrict__ p, int n) {
    int i = blockIdx.x * blockDim.x + threadIdx.x;
    if (i < n) p[i] = 0.f;
}

// transpose + f16-convert the 12 [128(k) x 128(n)] weight matrices into Bt[n][k]
__global__ __launch_bounds__(256) void k_w_prep(
    const float* __restrict__ Wnb, const float* __restrict__ Wnode,
    const float* __restrict__ Wm1, const float* __restrict__ Wm2,
    _Float16* __restrict__ wt) {
    int idx = blockIdx.x * blockDim.x + threadIdx.x;
    if (idx >= 12 * EMB * EMB) return;
    int m = idx >> 14;            // matrix id 0..11
    int r = idx & 16383;
    int n = r >> 7;               // out channel
    int k = r & 127;              // in channel
    int layer = m >> 2, which = m & 3;
    const float* src =
        (which == 0 ? Wnb : which == 1 ? Wnode : which == 2 ? Wm1 : Wm2) + layer * EMB * EMB;
    wt[idx] = (_Float16)src[k * EMB + n];   // dst flat = m*16384 + n*128 + k
}

// initial node MLP: t = node_attr(2) @ W0(2x128) + b0
__global__ __launch_bounds__(256) void k_init_mm(
    const float* __restrict__ na, const float* __restrict__ W0,
    const float* __restrict__ b0, float* __restrict__ out) {
    int idx = blockIdx.x * blockDim.x + threadIdx.x;
    if (idx >= TOT) return;
    int n = idx >> 7, ch = idx & 127;
    out[idx] = na[n * 2] * W0[ch] + na[n * 2 + 1] * W0[EMB + ch] + b0[ch];
}

// per-channel sum / sumsq over nodes (for BN stats)
__global__ __launch_bounds__(128) void k_stats(
    const float* __restrict__ z, float* __restrict__ sums,
    float* __restrict__ sumsq, int n_rows) {
    int ch = threadIdx.x;
    float s = 0.f, q = 0.f;
    for (int r = blockIdx.x; r < n_rows; r += gridDim.x) {
        float v = z[(size_t)r * EMB + ch];
        s += v; q += v * v;
    }
    unsafeAtomicAdd(&sums[ch], s);
    unsafeAtomicAdd(&sumsq[ch], q);
}

// BN fold: a = g*rsqrt(var+eps), sh = beta - mean*a
__global__ __launch_bounds__(128) void k_affine(
    const float* __restrict__ sums, const float* __restrict__ sumsq,
    const float* __restrict__ g, const float* __restrict__ bt,
    float* __restrict__ a, float* __restrict__ sh, float invn) {
    int ch = threadIdx.x;
    float m = sums[ch] * invn;
    float v = sumsq[ch] * invn - m * m;
    float s = g[ch] * rsqrtf(v + EPS);
    a[ch] = s;
    sh[ch] = bt[ch] - m * s;
}

// out = relu(z*a+sh) -> f16 (and optional f32 for final output)
__global__ __launch_bounds__(256) void k_bnrelu(
    const float* __restrict__ z, const float* __restrict__ a,
    const float* __restrict__ sh, _Float16* __restrict__ oh,
    float* __restrict__ of) {
    int idx = blockIdx.x * blockDim.x + threadIdx.x;
    if (idx >= TOT) return;
    int ch = idx & 127;
    float v = fmaxf(z[idx] * a[ch] + sh[ch], 0.f);
    oh[idx] = (_Float16)v;
    if (of) of[idx] = v;
}

// edge branch scalar accumulation: s[row] += e, c[row] += 1
__global__ __launch_bounds__(256) void k_edge_s(
    const float* __restrict__ ea, const int* __restrict__ row,
    float* __restrict__ sN, float* __restrict__ cN, int ne) {
    int e = blockIdx.x * blockDim.x + threadIdx.x;
    if (e >= ne) return;
    int r = row[e];
    unsafeAtomicAdd(&sN[r], ea[e]);
    unsafeAtomicAdd(&cN[r], 1.0f);
}

// 5 scalar moments over nodes: {Σs, Σs², Σc, Σc², Σsc}
__global__ __launch_bounds__(256) void k_scal_stats(
    const float* __restrict__ sN, const float* __restrict__ cN,
    float* __restrict__ scal, int n) {
    float s = 0.f, ss = 0.f, c = 0.f, cc = 0.f, sc = 0.f;
    for (int i = blockIdx.x * blockDim.x + threadIdx.x; i < n;
         i += blockDim.x * gridDim.x) {
        float a = sN[i], b = cN[i];
        s += a; ss += a * a; c += b; cc += b * b; sc += a * b;
    }
    unsafeAtomicAdd(&scal[0], s);  unsafeAtomicAdd(&scal[1], ss);
    unsafeAtomicAdd(&scal[2], c);  unsafeAtomicAdd(&scal[3], cc);
    unsafeAtomicAdd(&scal[4], sc);
}

// closed-form BN for the edge branch: ea_bn[n,ch] = s[n]*P + c[n]*Q + R
__global__ __launch_bounds__(128) void k_edge_affine(
    const float* __restrict__ scal, const float* __restrict__ We,
    const float* __restrict__ be, const float* __restrict__ ge,
    const float* __restrict__ bte, float* __restrict__ P,
    float* __restrict__ Q, float* __restrict__ R, float invn) {
    int ch = threadIdx.x;
    float ms = scal[0] * invn, vs = scal[1] * invn - ms * ms;
    float mc = scal[2] * invn, vc = scal[3] * invn - mc * mc;
    float cov = scal[4] * invn - ms * mc;
    float w = We[ch], b = be[ch];
    float mean = ms * w + mc * b;
    float var = w * w * vs + b * b * vc + 2.f * w * b * cov;
    float a = ge[ch] * rsqrtf(var + EPS);
    P[ch] = w * a; Q[ch] = b * a; R[ch] = bte[ch] - mean * a;
}

// neighbor scatter: aggr[col[e]] += H[row[e]], 4 channels/lane, 32 lanes/edge
__global__ __launch_bounds__(256) void k_scatter(
    const float* __restrict__ H, const int* __restrict__ row,
    const int* __restrict__ col, float* __restrict__ aggr, int ne) {
    int t = blockIdx.x * blockDim.x + threadIdx.x;
    int e = t >> 5;
    if (e >= ne) return;
    int lane = t & 31;
    int r = row[e], c = col[e];
    const float4 v = *(const float4*)(H + (size_t)r * EMB + lane * 4);
    float* dst = aggr + (size_t)c * EMB + lane * 4;
    unsafeAtomicAdd(dst + 0, v.x);
    unsafeAtomicAdd(dst + 1, v.y);
    unsafeAtomicAdd(dst + 2, v.z);
    unsafeAtomicAdd(dst + 3, v.w);
}

// y1 = relu(xn_bn + aggr_bn + ea_bn)  -> f16
__global__ __launch_bounds__(256) void k_combine(
    const float* __restrict__ xn, const float* __restrict__ aggr,
    const float* __restrict__ sN, const float* __restrict__ cN,
    const float* __restrict__ a0, const float* __restrict__ s0,
    const float* __restrict__ a1, const float* __restrict__ s1,
    const float* __restrict__ P, const float* __restrict__ Q,
    const float* __restrict__ R, _Float16* __restrict__ yh) {
    int idx = blockIdx.x * blockDim.x + threadIdx.x;
    if (idx >= TOT) return;
    int ch = idx & 127, n = idx >> 7;
    float v = xn[idx] * a0[ch] + s0[ch]
            + aggr[idx] * a1[ch] + s1[ch]
            + sN[n] * P[ch] + cN[n] * Q[ch] + R[ch];
    yh[idx] = (_Float16)fmaxf(v, 0.f);
}

// ---------------------------------------------------------------- WMMA GEMM
// C[40000x128] = A_h[40000x128] @ W[128x128] + bias, with Bt[n][k] = W[k][n] (f16).
// Block: 128 threads (4 waves), 16 rows x 128 cols. Wave w: cols [32w, 32w+32).
__global__ __launch_bounds__(128) void k_gemm(
    const _Float16* __restrict__ A, const _Float16* __restrict__ Bt,
    const float* __restrict__ bias, float* __restrict__ C) {
    const int lane = threadIdx.x & 31;
    const int wave = threadIdx.x >> 5;
    const int m0 = blockIdx.x << 4;
    const int mr = lane & 15;
    const int kb = (lane >> 4) << 3;      // 0 or 8 (K-half select per 16-bit A layout)
    const int n0 = wave << 5;

    const _Float16* ap  = A  + (size_t)(m0 + mr) * EMB;
    const _Float16* bp0 = Bt + (size_t)(n0 + mr) * EMB;
    const _Float16* bp1 = Bt + (size_t)(n0 + 16 + mr) * EMB;

    v8f acc0 = {0.f, 0.f, 0.f, 0.f, 0.f, 0.f, 0.f, 0.f};
    v8f acc1 = {0.f, 0.f, 0.f, 0.f, 0.f, 0.f, 0.f, 0.f};

#pragma unroll
    for (int k0 = 0; k0 < EMB; k0 += 32) {
        Frag a, b0, b1;
        a.h[0]  = *(const v8h*)(ap  + k0 + kb);
        a.h[1]  = *(const v8h*)(ap  + k0 + 16 + kb);
        b0.h[0] = *(const v8h*)(bp0 + k0 + kb);
        b0.h[1] = *(const v8h*)(bp0 + k0 + 16 + kb);
        b1.h[0] = *(const v8h*)(bp1 + k0 + kb);
        b1.h[1] = *(const v8h*)(bp1 + k0 + 16 + kb);
        acc0 = __builtin_amdgcn_wmma_f32_16x16x32_f16(false, a.v, false, b0.v,
                                                      (short)0, acc0, false, false);
        acc1 = __builtin_amdgcn_wmma_f32_16x16x32_f16(false, a.v, false, b1.v,
                                                      (short)0, acc1, false, false);
    }

    // C/D layout: VGPR r -> row m0 + r + 8*(lane>=16), col = n-tile + lane%16
    const int rb = (lane >> 4) << 3;
    const int c0 = n0 + (lane & 15);
    const int c1 = c0 + 16;
    const float bb0 = bias[c0], bb1 = bias[c1];
#pragma unroll
    for (int r = 0; r < 8; ++r) {
        const size_t offr = (size_t)(m0 + rb + r) * EMB;
        C[offr + c0] = acc0[r] + bb0;
        C[offr + c1] = acc1[r] + bb1;
    }
}

// ---------------------------------------------------------------- launcher
extern "C" void kernel_launch(void* const* d_in, const int* in_sizes, int n_in,
                              void* d_out, int out_size, void* d_ws, size_t ws_size,
                              hipStream_t stream) {
    (void)in_sizes; (void)n_in; (void)out_size; (void)ws_size;
    const float* node_attr = (const float*)d_in[0];
    const int*   eidx      = (const int*)d_in[1];
    const float* edge_attr = (const float*)d_in[2];
    const float* W0  = (const float*)d_in[3];
    const float* b0  = (const float*)d_in[4];
    const float* g0  = (const float*)d_in[5];
    const float* bt0 = (const float*)d_in[6];
    const float* Wnode = (const float*)d_in[7];
    const float* bnode = (const float*)d_in[8];
    const float* Wedge = (const float*)d_in[9];
    const float* bedge = (const float*)d_in[10];
    const float* Wnb = (const float*)d_in[11];
    const float* bnb = (const float*)d_in[12];
    const float* gn  = (const float*)d_in[13];
    const float* ge  = (const float*)d_in[14];
    const float* gnb = (const float*)d_in[15];
    const float* gm1 = (const float*)d_in[16];
    const float* gm2 = (const float*)d_in[17];
    const float* btn  = (const float*)d_in[18];
    const float* bte  = (const float*)d_in[19];
    const float* btnb = (const float*)d_in[20];
    const float* btm1 = (const float*)d_in[21];
    const float* btm2 = (const float*)d_in[22];
    const float* Wm1 = (const float*)d_in[23];
    const float* bm1 = (const float*)d_in[24];
    const float* Wm2 = (const float*)d_in[25];
    const float* bm2 = (const float*)d_in[26];

    const int* row = eidx;
    const int* col = eidx + NE;

    char* ws = (char*)d_ws;
    size_t off = 0;
    auto take = [&](size_t bytes) -> void* {
        void* p = ws + off;
        off = (off + bytes + 255) & ~(size_t)255;
        return p;
    };
    _Float16* wt   = (_Float16*)take((size_t)12 * EMB * EMB * 2);
    _Float16* xh   = (_Float16*)take((size_t)TOT * 2);
    _Float16* yh   = (_Float16*)take((size_t)TOT * 2);
    float*    bufA = (float*)take((size_t)TOT * 4);
    float*    aggr = (float*)take((size_t)TOT * 4);
    float*    sN   = (float*)take((size_t)NN * 4);
    float*    cN   = (float*)take((size_t)NN * 4);
    float*    sml  = (float*)take(2048 * 4);
    float* sums  = sml;        float* sumsq = sml + 128;
    float* a0    = sml + 256;  float* s0    = sml + 384;
    float* a1    = sml + 512;  float* s1    = sml + 640;
    float* P     = sml + 768;  float* Q     = sml + 896;  float* R = sml + 1024;
    float* scal  = sml + 1152;

    const float invN = 1.0f / (float)NN;
    const int BLK_TOT = (TOT + 255) / 256;          // 20000
    const int BLK_NN  = (NN + 255) / 256;

    // ---- one-time prep (per call, deterministic)
    k_w_prep<<<(12 * EMB * EMB + 255) / 256, 256, 0, stream>>>(Wnb, Wnode, Wm1, Wm2, wt);
    k_zero<<<BLK_NN, 256, 0, stream>>>(sN, NN);
    k_zero<<<BLK_NN, 256, 0, stream>>>(cN, NN);
    k_zero<<<1, 32, 0, stream>>>(scal, 8);
    k_edge_s<<<(NE + 255) / 256, 256, 0, stream>>>(edge_attr, row, sN, cN, NE);
    k_scal_stats<<<80, 256, 0, stream>>>(sN, cN, scal, NN);

    // ---- initial node MLP + BN + ReLU -> xh (f16)
    k_init_mm<<<BLK_TOT, 256, 0, stream>>>(node_attr, W0, b0, bufA);
    k_zero<<<1, 256, 0, stream>>>(sums, 256);
    k_stats<<<256, 128, 0, stream>>>(bufA, sums, sumsq, NN);
    k_affine<<<1, 128, 0, stream>>>(sums, sumsq, g0, bt0, a0, s0, invN);
    k_bnrelu<<<BLK_TOT, 256, 0, stream>>>(bufA, a0, s0, xh, (float*)nullptr);

    for (int i = 0; i < 3; ++i) {
        const _Float16* wnb = wt + (size_t)(i * 4 + 0) * EMB * EMB;
        const _Float16* wnd = wt + (size_t)(i * 4 + 1) * EMB * EMB;
        const _Float16* wt1 = wt + (size_t)(i * 4 + 2) * EMB * EMB;
        const _Float16* wt2 = wt + (size_t)(i * 4 + 3) * EMB * EMB;

        k_edge_affine<<<1, 128, 0, stream>>>(scal, Wedge + i * EMB, bedge + i * EMB,
                                             ge + i * EMB, bte + i * EMB, P, Q, R, invN);

        // neighbor branch: H = x @ Wnb + bnb ; scatter ; BN stats
        k_gemm<<<NN / 16, 128, 0, stream>>>(xh, wnb, bnb + i * EMB, bufA);
        k_zero<<<BLK_TOT, 256, 0, stream>>>(aggr, TOT);
        k_scatter<<<(NE * 32) / 256, 256, 0, stream>>>(bufA, row, col, aggr, NE);
        k_zero<<<1, 256, 0, stream>>>(sums, 256);
        k_stats<<<256, 128, 0, stream>>>(aggr, sums, sumsq, NN);
        k_affine<<<1, 128, 0, stream>>>(sums, sumsq, gnb + i * EMB, btnb + i * EMB, a1, s1, invN);

        // node branch: XN = x @ Wnode + bnode ; BN stats
        k_gemm<<<NN / 16, 128, 0, stream>>>(xh, wnd, bnode + i * EMB, bufA);
        k_zero<<<1, 256, 0, stream>>>(sums, 256);
        k_stats<<<256, 128, 0, stream>>>(bufA, sums, sumsq, NN);
        k_affine<<<1, 128, 0, stream>>>(sums, sumsq, gn + i * EMB, btn + i * EMB, a0, s0, invN);

        // y1 = relu(xn_bn + aggr_bn + ea_bn) -> yh (f16)
        k_combine<<<BLK_TOT, 256, 0, stream>>>(bufA, aggr, sN, cN,
                                               a0, s0, a1, s1, P, Q, R, yh);

        // y2 = relu(bn(y1 @ Wm1 + bm1)) -> xh (f16)
        k_gemm<<<NN / 16, 128, 0, stream>>>(yh, wt1, bm1 + i * EMB, bufA);
        k_zero<<<1, 256, 0, stream>>>(sums, 256);
        k_stats<<<256, 128, 0, stream>>>(bufA, sums, sumsq, NN);
        k_affine<<<1, 128, 0, stream>>>(sums, sumsq, gm1 + i * EMB, btm1 + i * EMB, a0, s0, invN);
        k_bnrelu<<<BLK_TOT, 256, 0, stream>>>(bufA, a0, s0, xh, (float*)nullptr);

        // x = relu(bn(y2 @ Wm2 + bm2)) -> xh (f16), f32 to d_out on last layer
        k_gemm<<<NN / 16, 128, 0, stream>>>(xh, wt2, bm2 + i * EMB, bufA);
        k_zero<<<1, 256, 0, stream>>>(sums, 256);
        k_stats<<<256, 128, 0, stream>>>(bufA, sums, sumsq, NN);
        k_affine<<<1, 128, 0, stream>>>(sums, sumsq, gm2 + i * EMB, btm2 + i * EMB, a0, s0, invN);
        k_bnrelu<<<BLK_TOT, 256, 0, stream>>>(bufA, a0, s0, xh,
                                              (i == 2) ? (float*)d_out : (float*)nullptr);
    }
}
